// RWKV_TimeMixing_84413287235982
// MI455X (gfx1250) — matrix operand order repaired and checked by reading, not compiled, not run on hardware
//
#include <hip/hip_runtime.h>
#include <stdint.h>
#include <stddef.h>

// ---------------- problem constants ----------------
#define B_ 8
#define T_ 4096
#define D_ 768
#define H_ 12
#define K_ 64
#define M_ (B_ * T_)          // 32768 rows
#define CHUNK 64              // scan chunk length
#define NCHUNK (T_ / CHUNK)   // 64 chunks
#define GN_SEG 16             // groupnorm partial segments per group

// ---------------- WMMA / TDM types (CDNA5 wave32) ----------------
typedef __attribute__((ext_vector_type(16))) __bf16 v16bf;
typedef __attribute__((ext_vector_type(8)))  float  v8f;
typedef __attribute__((ext_vector_type(4)))  unsigned int v4u;
typedef __attribute__((ext_vector_type(8)))  int    v8i_;
typedef __attribute__((ext_vector_type(4)))  int    v4i_;

union Frag16 { v16bf v; uint32_t u[8]; };

__device__ __forceinline__ uint16_t f2bf(float f) {
  // round-to-nearest-even fp32 -> bf16
  uint32_t u = __float_as_uint(f);
  uint32_t r = (u + 0x7FFFu + ((u >> 16) & 1u)) >> 16;
  return (uint16_t)r;
}

// ---- CDNA5 async global->LDS copy (ASYNCcnt-tracked, GV addressing) ----
__device__ __forceinline__ void async_cp16(uint32_t lds_addr, const uint16_t* g) {
  asm volatile("global_load_async_to_lds_b128 %0, %1, off"
               :: "v"(lds_addr), "v"(g)
               : "memory");
}
__device__ __forceinline__ void wait_async2() {
  asm volatile("s_wait_asynccnt 0x2" ::: "memory");
}
__device__ __forceinline__ void wait_async0() {
  asm volatile("s_wait_asynccnt 0x0" ::: "memory");
}
// low 32 bits of a generic pointer to __shared__ == LDS byte address
__device__ __forceinline__ uint32_t lds_addr_of(const void* p) {
  return (uint32_t)(uintptr_t)p;
}

// ---- CDNA5 Tensor Data Mover: 2-D W tile (64 rows x 32 bf16) -> LDS ----
// D# group0: count=1(user), lds_addr, global_addr, type=2("image").
// D# group1: data_size=2B, pad_enable, pad_interval=16 DW (one 64B row),
//            pad_amount=4 DW (16B) -> LDS row stride 80B; tensor 768x768,
//            tile 32(k) x 64(n), dim0 stride 768.
__device__ __forceinline__ void tdm_load_w_tile(uint32_t lds_byte,
                                                const uint16_t* gsrc) {
  uint64_t ga = (uint64_t)(uintptr_t)gsrc;
  v4u g0;
  g0[0] = 1u;                                       // count=1, user desc
  g0[1] = lds_byte;                                 // lds_addr
  g0[2] = (uint32_t)ga;                             // global_addr[31:0]
  g0[3] = (uint32_t)((ga >> 32) & 0x1FFFFFFu)       // global_addr[56:32]
        | (2u << 30);                               // type = 2
  v8i_ g1;
  g1[0] = (1 << 16)                                 // data_size = 2B
        | (1 << 20)                                 // pad_enable
        | (3 << 22)                                 // pad_interval: 16 DWORDs
        | (3 << 25);                                // pad_amount: 4 DWORDs
  g1[1] = (int)(768u << 16);                        // tensor_dim0 = 768
  g1[2] = (int)(768u << 16);                        // tensor_dim1 = 768
  g1[3] = (int)(32u << 16);                         // tile_dim0 = 32 (k)
  g1[4] = 64;                                       // tile_dim1 = 64 (n rows)
  g1[5] = 768;                                      // tensor_dim0_stride = 768
  g1[6] = 0;
  g1[7] = 0;
  v4i_ gz = {0, 0, 0, 0};                           // 2-D: groups 2/3 unused
#if defined(__clang_major__) && (__clang_major__ >= 23)
  v8i_ gz8 = {0, 0, 0, 0, 0, 0, 0, 0};
  __builtin_amdgcn_tensor_load_to_lds(g0, g1, gz, gz, gz8, 0);
#else
  __builtin_amdgcn_tensor_load_to_lds(g0, g1, gz, gz, 0);
#endif
}
__device__ __forceinline__ void wait_tensor1() {
  __builtin_amdgcn_s_wait_tensorcnt(1);
}
__device__ __forceinline__ void wait_tensor0() {
  __builtin_amdgcn_s_wait_tensorcnt(0);
}

// ---------------- elementwise: f32 -> bf16 convert ----------------
__global__ __launch_bounds__(256) void conv_bf16_kernel(
    const float* __restrict__ in, uint16_t* __restrict__ out, int n) {
  int i = blockIdx.x * 256 + threadIdx.x;
  if (i < n) out[i] = f2bf(in[i]);
}

// ---------------- precompute w = exp(-exp(td)), eu = exp(tf), wL = w^CHUNK ----
__global__ __launch_bounds__(256) void params_kernel(
    const float* __restrict__ td, const float* __restrict__ tf,
    float* __restrict__ wdec, float* __restrict__ eu, float* __restrict__ wl) {
  int i = blockIdx.x * 256 + threadIdx.x;
  if (i < D_) {
    float e = expf(td[i]);
    wdec[i] = expf(-e);
    wl[i]   = expf(-(float)CHUNK * e);   // w^CHUNK
    eu[i]   = expf(tf[i]);
  }
}

// ---------------- token-shift mix -> bf16 (float4 per thread) ----------------
__global__ __launch_bounds__(256) void mix_kernel(
    const float* __restrict__ x,
    const float* __restrict__ tmr, const float* __restrict__ tmk,
    const float* __restrict__ tmv,
    uint16_t* __restrict__ xr, uint16_t* __restrict__ xk,
    uint16_t* __restrict__ xv) {
  size_t i4  = (size_t)blockIdx.x * 256 + threadIdx.x;   // float4 index
  size_t idx = i4 * 4;
  int d      = (int)(idx % D_);
  size_t bt  = idx / D_;
  int t      = (int)(bt % T_);
  const float4 xc = ((const float4*)x)[i4];
  float4 xp = make_float4(0.f, 0.f, 0.f, 0.f);
  if (t != 0) xp = ((const float4*)x)[i4 - D_ / 4];
#pragma unroll
  for (int q = 0; q < 4; ++q) {
    float c = (&xc.x)[q], p = (&xp.x)[q];
    float mr = tmr[d + q], mk = tmk[d + q], mv = tmv[d + q];
    xr[idx + q] = f2bf(c * mr + p * (1.f - mr));
    xk[idx + q] = f2bf(c * mk + p * (1.f - mk));
    xv[idx + q] = f2bf(c * mv + p * (1.f - mv));
  }
}

// ---------------- bf16 WMMA GEMM: C[M,768] = act(A[M,768] @ W[768,768]^T) ----
// Block tile 128(M) x 64(N), K-step 32; 256 threads = 8 wave32s (4x2 wave grid),
// each wave computes 2x2 WMMA 16x16 tiles via v_wmma_f32_16x16x32_bf16.
// A tiles: GLOBAL_LOAD_ASYNC_TO_LDS_B128 (2 per thread per tile, ASYNCcnt).
// W tiles: Tensor Data Mover descriptor load issued by wave 0 (TENSORcnt),
// with LDS row padding done by the TDM. Double-buffered LDS; issue tile i+1
// then wait asynccnt<=2 / tensorcnt<=1 -> tile i resident (in-order engines).
// ACT: 0 = identity, 1 = sigmoid.
template <int ACT>
__global__ __launch_bounds__(256) void gemm_bf16_kernel(
    const uint16_t* __restrict__ A,   // bf16 [M, 768] row-major
    const uint16_t* __restrict__ W,   // bf16 [768, 768] row-major ([out, in])
    float* __restrict__ C) {          // f32  [M, 768]
  constexpr int Dk   = D_;
  constexpr int LDSW = 20;            // uint32 per row (40 halves, padded)
  constexpr int NK   = Dk / 32;       // 24 K-tiles
  __shared__ uint32_t lds_a[2][128 * LDSW];
  __shared__ uint32_t lds_b[2][64 * LDSW];

  const int tid  = threadIdx.x;
  const int lane = tid & 31;
  const int wid  = tid >> 5;
  const int wm   = wid & 3;           // wave M position (0..3) -> 32 rows each
  const int wn   = wid >> 2;          // wave N position (0..1) -> 32 cols each
  const int m_base = blockIdx.y * 128;
  const int n_base = blockIdx.x * 64;

  // per-thread A staging coordinates (two 16B granules per half-row)
  const int arow = tid >> 1;
  const int ac0  = (tid & 1);
  const int ac1  = (tid & 1) + 2;
  const uint16_t* gA0 = A + (size_t)(m_base + arow) * Dk + ac0 * 8;
  const uint16_t* gA1 = A + (size_t)(m_base + arow) * Dk + ac1 * 8;
  const uint16_t* gW  = W + (size_t)n_base * Dk;    // TDM tile base (uniform)

  v8f acc[2][2] = {};                 // zero-initialized accumulators

  auto issue_tile = [&](int k0, int buf) {
    async_cp16(lds_addr_of(&lds_a[buf][arow * LDSW + ac0 * 4]), gA0 + k0);
    async_cp16(lds_addr_of(&lds_a[buf][arow * LDSW + ac1 * 4]), gA1 + k0);
    if (wid == 0)
      tdm_load_w_tile(lds_addr_of(&lds_b[buf][0]), gW + k0);
  };

  issue_tile(0, 0);

#pragma unroll 2
  for (int it = 0; it < NK; ++it) {
    const int buf = it & 1;
    if (it + 1 < NK) {
      issue_tile((it + 1) * 32, buf ^ 1);
      // L2 prefetch two tiles ahead of the async/TDM engines
      if (it + 2 < NK)
        __builtin_prefetch(gA0 + (it + 2) * 32, 0, 3);
      wait_async2();                  // this thread's tile-`it` A copies landed
      if (wid == 0) wait_tensor1();   // tile-`it` TDM landed
    } else {
      wait_async0();
      if (wid == 0) wait_tensor0();
    }
    __syncthreads();

    // ---- gather WMMA fragments per ISA VGPR layouts ----
    Frag16 afr[2], bfr[2];
    const int mrow = lane & 15;
    const int hi   = lane >> 4;
#pragma unroll
    for (int i = 0; i < 2; ++i) {
      int rbase = (wm * 32 + i * 16 + mrow) * LDSW;
#pragma unroll
      for (int j = 0; j < 8; ++j) {
        // A 16x32 bf16 layout: lanes 0-15 K 0..7 / 16..23, lanes 16-31 K 8..15 / 24..31
        int kk = 2 * (j & 3) + 16 * (j >> 2) + 8 * hi;
        afr[i].u[j] = lds_a[buf][rbase + (kk >> 1)];
      }
    }
#pragma unroll
    for (int j2 = 0; j2 < 2; ++j2) {
      int rbase = (wn * 32 + j2 * 16 + mrow) * LDSW;
#pragma unroll
      for (int j = 0; j < 8; ++j) {
        // B 32x16 bf16 layout: lanes 0-15 K 0..7 / 16..23, lanes 16-31 K 8..15 / 24..31
        int kk = 2 * (j & 3) + 16 * (j >> 2) + 8 * hi;
        bfr[j2].u[j] = lds_b[buf][rbase + (kk >> 1)];
      }
    }

#pragma unroll
    for (int i = 0; i < 2; ++i)
#pragma unroll
      for (int j2 = 0; j2 < 2; ++j2)
        acc[i][j2] = __builtin_amdgcn_wmma_f32_16x16x32_bf16(
            false, afr[i].v, false, bfr[j2].v, (short)0, acc[i][j2],
            false, false);
    __syncthreads();   // all waves done reading buf -> safe to overwrite
  }

  // ---- epilogue: C/D layout VGPR r, lanes 0-15 -> M=r, lanes 16-31 -> M=8+r ----
  const int nlane = lane & 15;
  const int mhalf = (lane >> 4) * 8;
#pragma unroll
  for (int i = 0; i < 2; ++i)
#pragma unroll
    for (int j2 = 0; j2 < 2; ++j2) {
      int n  = n_base + wn * 32 + j2 * 16 + nlane;
      int mb = m_base + wm * 32 + i * 16 + mhalf;
#pragma unroll
      for (int rr = 0; rr < 8; ++rr) {
        float val = acc[i][j2][rr];
        if (ACT == 1) val = 1.f / (1.f + expf(-val));
        C[(size_t)(mb + rr) * D_ + n] = val;
      }
    }
}

// ---------------- WKV chunked scan ----------------
// Pass A: local chunk sums (zero-init):  S = sum_i w^(L-1-i) * e^{k_i} * {v_i, 1}
__global__ __launch_bounds__(256) void wkv_local_kernel(
    const float* __restrict__ kf, const float* __restrict__ vf,
    const float* __restrict__ wdec,
    float* __restrict__ Snum, float* __restrict__ Sden) {
  int id   = blockIdx.x * 256 + threadIdx.x;  // b * NCHUNK * D_ threads
  int ch   = id % D_;
  int rest = id / D_;
  int c    = rest % NCHUNK;
  int b    = rest / NCHUNK;
  float w  = wdec[ch];
  float num = 0.f, den = 0.f;
  size_t base = ((size_t)(b * T_ + c * CHUNK)) * D_ + ch;
#pragma unroll 4
  for (int t = 0; t < CHUNK; ++t) {
    float ek = expf(kf[base]);
    float vv = vf[base];
    num = num * w + ek * vv;
    den = den * w + ek;
    base += D_;
  }
  int sidx = (b * NCHUNK + c) * D_ + ch;
  Snum[sidx] = num;
  Sden[sidx] = den;
}

// Pass B: exclusive cross-chunk scan with transform s' = w^L * s + S_c
__global__ __launch_bounds__(256) void wkv_prefix_kernel(
    const float* __restrict__ Snum, const float* __restrict__ Sden,
    const float* __restrict__ wl,
    float* __restrict__ Pnum, float* __restrict__ Pden) {
  int id = blockIdx.x * 256 + threadIdx.x;   // B_*D_ threads
  if (id >= B_ * D_) return;
  int ch = id % D_;
  int b  = id / D_;
  float w = wl[ch];
  float num = 0.f, den = 0.f;
  for (int c = 0; c < NCHUNK; ++c) {
    int sidx = (b * NCHUNK + c) * D_ + ch;
    Pnum[sidx] = num;
    Pden[sidx] = den;
    num = num * w + Snum[sidx];
    den = den * w + Sden[sidx];
  }
}

// Pass C: replay chunk from prefix state, emit y = r * wkv
__global__ __launch_bounds__(256) void wkv_apply_kernel(
    const float* __restrict__ rf, const float* __restrict__ kf,
    const float* __restrict__ vf, const float* __restrict__ wdec,
    const float* __restrict__ eu,
    const float* __restrict__ Pnum, const float* __restrict__ Pden,
    float* __restrict__ y) {
  int id   = blockIdx.x * 256 + threadIdx.x;
  int ch   = id % D_;
  int rest = id / D_;
  int c    = rest % NCHUNK;
  int b    = rest / NCHUNK;
  int sidx = (b * NCHUNK + c) * D_ + ch;
  float w   = wdec[ch];
  float u   = eu[ch];
  float num = Pnum[sidx];
  float den = Pden[sidx];
  size_t base = ((size_t)(b * T_ + c * CHUNK)) * D_ + ch;
#pragma unroll 4
  for (int t = 0; t < CHUNK; ++t) {
    float ek = expf(kf[base]);
    float vv = vf[base];
    float wkv = (num + u * ek * vv) / (den + u * ek + 1e-9f);
    y[base] = rf[base] * wkv;
    num = num * w + ek * vv;
    den = den * w + ek;
    base += D_;
  }
}

// ---------------- GroupNorm: two-stage reduce over (T, K) per (b, h) ----------
__global__ __launch_bounds__(256) void gn_partial_kernel(
    const float* __restrict__ y, float* __restrict__ psum,
    float* __restrict__ psq) {
  __shared__ float s1[256];
  __shared__ float s2[256];
  int g   = blockIdx.x;          // 0..B_*H_-1
  int seg = blockIdx.y;          // 0..GN_SEG-1
  int b = g / H_, h = g % H_;
  const int per = (T_ * K_) / GN_SEG;   // 16384 elements per segment
  float sum = 0.f, sq = 0.f;
  for (int i = threadIdx.x; i < per; i += 256) {
    int e = seg * per + i;
    int t = e >> 6, kc = e & 63;
    float val = y[((size_t)(b * T_ + t)) * D_ + h * K_ + kc];
    sum += val;
    sq  += val * val;
  }
  s1[threadIdx.x] = sum;
  s2[threadIdx.x] = sq;
  __syncthreads();
  for (int off = 128; off > 0; off >>= 1) {
    if (threadIdx.x < off) {
      s1[threadIdx.x] += s1[threadIdx.x + off];
      s2[threadIdx.x] += s2[threadIdx.x + off];
    }
    __syncthreads();
  }
  if (threadIdx.x == 0) {
    psum[g * GN_SEG + seg] = s1[0];
    psq [g * GN_SEG + seg] = s2[0];
  }
}

__global__ __launch_bounds__(128) void gn_final_kernel(
    const float* __restrict__ psum, const float* __restrict__ psq,
    float* __restrict__ mean, float* __restrict__ rstd) {
  int g = blockIdx.x * 128 + threadIdx.x;
  if (g >= B_ * H_) return;
  float s = 0.f, q = 0.f;
#pragma unroll
  for (int i = 0; i < GN_SEG; ++i) {
    s += psum[g * GN_SEG + i];
    q += psq [g * GN_SEG + i];
  }
  const float inv = 1.f / (float)(T_ * K_);
  float m   = s * inv;
  float var = q * inv - m * m;
  mean[g] = m;
  rstd[g] = rsqrtf(var + 1e-5f);
}

// normalize + affine + bf16 convert (feeds the W_o GEMM)
__global__ __launch_bounds__(256) void gn_norm_kernel(
    const float* __restrict__ y, const float* __restrict__ mean,
    const float* __restrict__ rstd, const float* __restrict__ gnw,
    const float* __restrict__ gnb, uint16_t* __restrict__ ybf) {
  size_t idx = (size_t)blockIdx.x * 256 + threadIdx.x;
  int d     = (int)(idx % D_);
  size_t bt = idx / D_;
  int b     = (int)(bt / T_);
  int g     = b * H_ + d / K_;
  float val = (y[idx] - mean[g]) * rstd[g] * gnw[d] + gnb[d];
  ybf[idx] = f2bf(val);
}

// ---------------- host-side orchestration ----------------
extern "C" void kernel_launch(void* const* d_in, const int* in_sizes, int n_in,
                              void* d_out, int out_size, void* d_ws, size_t ws_size,
                              hipStream_t stream) {
  const float* x   = (const float*)d_in[0];
  const float* tmr = (const float*)d_in[1];
  const float* tmk = (const float*)d_in[2];
  const float* tmv = (const float*)d_in[3];
  const float* td  = (const float*)d_in[4];
  const float* tf  = (const float*)d_in[5];
  const float* Wr  = (const float*)d_in[6];
  const float* Wk  = (const float*)d_in[7];
  const float* Wv  = (const float*)d_in[8];
  const float* Wo  = (const float*)d_in[9];
  const float* gnw = (const float*)d_in[10];
  const float* gnb = (const float*)d_in[11];
  float* out = (float*)d_out;

  // ---- workspace layout (256B aligned slabs) ----
  char* ws = (char*)d_ws;
  size_t off = 0;
  auto take = [&](size_t bytes) -> char* {
    char* p = ws + off;
    off = (off + bytes + 255) & ~(size_t)255;
    return p;
  };
  uint16_t* Wr_bf = (uint16_t*)take((size_t)D_ * D_ * 2);
  uint16_t* Wk_bf = (uint16_t*)take((size_t)D_ * D_ * 2);
  uint16_t* Wv_bf = (uint16_t*)take((size_t)D_ * D_ * 2);
  uint16_t* Wo_bf = (uint16_t*)take((size_t)D_ * D_ * 2);
  float* wdec = (float*)take(D_ * 4);
  float* eu   = (float*)take(D_ * 4);
  float* wl   = (float*)take(D_ * 4);
  float* mean = (float*)take(B_ * H_ * 4);
  float* rstd = (float*)take(B_ * H_ * 4);
  float* psum = (float*)take(B_ * H_ * GN_SEG * 4);
  float* psq  = (float*)take(B_ * H_ * GN_SEG * 4);
  // bf16 mix buffers; reused after the r/k/v GEMMs for y(f32) + y(bf16)
  char* Abuf = take((size_t)3 * M_ * D_ * 2);
  uint16_t* xr_bf = (uint16_t*)Abuf;
  uint16_t* xk_bf = (uint16_t*)(Abuf + (size_t)M_ * D_ * 2);
  uint16_t* xv_bf = (uint16_t*)(Abuf + (size_t)2 * M_ * D_ * 2);
  float*    y_f   = (float*)Abuf;                                  // M*D*4 bytes
  uint16_t* y_bf  = (uint16_t*)(Abuf + (size_t)M_ * D_ * 4);       // M*D*2 bytes
  float* r_f = (float*)take((size_t)M_ * D_ * 4);
  float* k_f = (float*)take((size_t)M_ * D_ * 4);
  float* v_f = (float*)take((size_t)M_ * D_ * 4);
  float* Snum = (float*)take((size_t)B_ * NCHUNK * D_ * 4);
  float* Sden = (float*)take((size_t)B_ * NCHUNK * D_ * 4);
  float* Pnum = (float*)take((size_t)B_ * NCHUNK * D_ * 4);
  float* Pden = (float*)take((size_t)B_ * NCHUNK * D_ * 4);
  (void)ws_size; (void)in_sizes; (void)n_in; (void)out_size;

  const int nW = D_ * D_;  // 589824

  // 1) weight + parameter precompute
  conv_bf16_kernel<<<nW / 256, 256, 0, stream>>>(Wr, Wr_bf, nW);
  conv_bf16_kernel<<<nW / 256, 256, 0, stream>>>(Wk, Wk_bf, nW);
  conv_bf16_kernel<<<nW / 256, 256, 0, stream>>>(Wv, Wv_bf, nW);
  conv_bf16_kernel<<<nW / 256, 256, 0, stream>>>(Wo, Wo_bf, nW);
  params_kernel<<<(D_ + 255) / 256, 256, 0, stream>>>(td, tf, wdec, eu, wl);

  // 2) token-shift mixes -> bf16
  mix_kernel<<<(M_ * D_ / 4) / 256, 256, 0, stream>>>(x, tmr, tmk, tmv,
                                                      xr_bf, xk_bf, xv_bf);

  // 3) projections (WMMA bf16, f32 accumulate; async-LDS + TDM staged)
  dim3 ggrid(D_ / 64, M_ / 128);
  gemm_bf16_kernel<1><<<ggrid, 256, 0, stream>>>(xr_bf, Wr_bf, r_f);  // sigmoid
  gemm_bf16_kernel<0><<<ggrid, 256, 0, stream>>>(xk_bf, Wk_bf, k_f);
  gemm_bf16_kernel<0><<<ggrid, 256, 0, stream>>>(xv_bf, Wv_bf, v_f);

  // 4) WKV chunked scan (linear recurrence -> 3-pass parallel scan)
  const int nScan = B_ * NCHUNK * D_;  // 393216
  wkv_local_kernel<<<nScan / 256, 256, 0, stream>>>(k_f, v_f, wdec, Snum, Sden);
  wkv_prefix_kernel<<<(B_ * D_ + 255) / 256, 256, 0, stream>>>(Snum, Sden, wl,
                                                               Pnum, Pden);
  wkv_apply_kernel<<<nScan / 256, 256, 0, stream>>>(r_f, k_f, v_f, wdec, eu,
                                                    Pnum, Pden, y_f);

  // 5) GroupNorm over (T, K) per (b, h); two-stage reduce, normalize -> bf16
  gn_partial_kernel<<<dim3(B_ * H_, GN_SEG), 256, 0, stream>>>(y_f, psum, psq);
  gn_final_kernel<<<1, 128, 0, stream>>>(psum, psq, mean, rstd);
  gn_norm_kernel<<<((size_t)M_ * D_) / 256, 256, 0, stream>>>(y_f, mean, rstd,
                                                              gnw, gnb, y_bf);

  // 6) output projection -> d_out (f32)
  gemm_bf16_kernel<0><<<ggrid, 256, 0, stream>>>(y_bf, Wo_bf, out);
}
